// bilstm_4037269258404
// MI455X (gfx1250) — matrix operand order repaired and checked by reading, not compiled, not run on hardware
//
#include <hip/hip_runtime.h>

// ---------------------------------------------------------------------------
// Bi-LSTM for MI455X (gfx1250): WMMA bf16 GEMMs + TDM-staged LDS-resident
// recurrent weights + persistent recurrent kernel with per-step barrier.
// B=32, T=2048, D=256, U=256.
// ---------------------------------------------------------------------------

typedef __attribute__((ext_vector_type(16))) __bf16 v16bf;
typedef __attribute__((ext_vector_type(8)))  __bf16 v8bf;
typedef __attribute__((ext_vector_type(8)))  float  v8f;
typedef __attribute__((ext_vector_type(4)))  float  v4f;
typedef __attribute__((ext_vector_type(4)))  unsigned int v4u;
typedef __attribute__((ext_vector_type(4)))  int    v4i;
typedef __attribute__((ext_vector_type(8)))  int    v8i;

#define T_ 2048
#define B_ 32
#define D_ 256
#define U_ 256
#define NWG_DIR 4          // workgroups per direction in recurrent kernel
#define PACK_ELEMS 262144  // bf16 elems per packed 256x1024 matrix (8kt*64nt*32lane*16)

__device__ __forceinline__ v8f wmma_bf16(v16bf a, v16bf b, v8f c) {
  // D = A(16x32 bf16) * B(32x16 bf16) + C(16x16 f32)
  return __builtin_amdgcn_wmma_f32_16x16x32_bf16(
      /*neg_a=*/false, a, /*neg_b=*/false, b,
      /*c_mod=*/(short)0, c, /*reuse_a=*/false, /*reuse_b=*/false);
}

__device__ __forceinline__ float sigm_f(float x) {
  return 1.0f / (1.0f + __expf(-x));
}
__device__ __forceinline__ float tanh_f(float x) {
  x = fminf(fmaxf(x, -15.0f), 15.0f);
  float e = __expf(2.0f * x);
  return (e - 1.0f) / (e + 1.0f);
}

// ---------------------------------------------------------------------------
// init: zero h double-buffers and barrier counters (must be reset every call)
// ---------------------------------------------------------------------------
__global__ void bilstm_init_kernel(__bf16* hbuf, unsigned* cnt) {
  int i = blockIdx.x * 256 + threadIdx.x;
  if (i < 2 * 2 * B_ * U_) hbuf[i] = (__bf16)0.0f;   // [parity][dir][32][256]
  if (i < 8) cnt[i] = 0u;
}

// ---------------------------------------------------------------------------
// pack: fp32 row-major [256][1024] -> bf16 WMMA B-operand tiles.
// Tile (kt,nt): lane L holds col n = nt*16+(L&15), k = kt*32+8*(L>>4)+e+(e>=8?8:0),
// e=0..15, stored contiguously (32B per lane).
// mat: 0=W_fw 1=W_bw 2=R_fw 3=R_bw
// ---------------------------------------------------------------------------
__global__ void bilstm_pack_kernel(const float* __restrict__ Wf,
                                   const float* __restrict__ Wb,
                                   const float* __restrict__ Rf,
                                   const float* __restrict__ Rb,
                                   __bf16* __restrict__ pW,
                                   __bf16* __restrict__ pR) {
  int gid  = blockIdx.x * 256 + threadIdx.x;   // 65536 threads
  int mat  = gid >> 14;
  int rem  = gid & 16383;                      // (kt*64+nt)*32 + lane
  int tile = rem >> 5;
  int l    = rem & 31;
  int kt = tile >> 6, nt = tile & 63;
  int n     = nt * 16 + (l & 15);
  int kbase = kt * 32 + 8 * (l >> 4);
  const float* src = (mat == 0) ? Wf : (mat == 1) ? Wb : (mat == 2) ? Rf : Rb;
  __bf16* dst = ((mat < 2) ? pW : pR) + (size_t)(mat & 1) * PACK_ELEMS
                + (size_t)rem * 16;
  v16bf v;
#pragma unroll
  for (int e = 0; e < 16; ++e)
    v[e] = (__bf16)src[(size_t)(kbase + e + (e >= 8 ? 8 : 0)) * 1024 + n];
  *(v16bf*)dst = v;
}

// ---------------------------------------------------------------------------
// projection: zx[dir] = x(_time-reversed for bw_) @ W[dir] + b[dir]
// M-dim remapped as m = t*32 + b so each 16x16 C tile is exactly one
// (t, batch-half, ntile) tile consumed by the recurrent kernel.
// Grid: idx = ((dir*2048 + t)*2 + mt)*2 + ntg ; 256 thr = 8 waves,
// wave handles 4 ntiles of one (dir,t,mt).
// ---------------------------------------------------------------------------
__global__ void __launch_bounds__(256)
bilstm_proj_kernel(const float* __restrict__ x, const __bf16* __restrict__ pW,
                   const float* __restrict__ bfw, const float* __restrict__ bbw,
                   __bf16* __restrict__ zxp) {
  int idx = blockIdx.x;
  int ntg = idx & 1;  idx >>= 1;
  int mt  = idx & 1;  idx >>= 1;
  int t   = idx & 2047;
  int dir = idx >> 11;

  int tid = threadIdx.x, lane = tid & 31, wave = tid >> 5;
  int ntbase = ntg * 32 + wave * 4;
  int half = lane >> 4;
  int b    = mt * 16 + (lane & 15);           // A-operand row (batch)
  int tt   = dir ? (2047 - t) : t;            // bw processes reversed time
  const float* bias = dir ? bbw : bfw;
  const __bf16* pWd = pW + (size_t)dir * PACK_ELEMS;
  const float* xrow = x + ((size_t)b * T_ + tt) * D_;

  v8f acc[4] = {};
#pragma unroll
  for (int kt = 0; kt < 8; ++kt) {
    int k0 = kt * 32 + 8 * half;
    v4f a0 = *(const v4f*)(xrow + k0);
    v4f a1 = *(const v4f*)(xrow + k0 + 4);
    v4f a2 = *(const v4f*)(xrow + k0 + 16);
    v4f a3 = *(const v4f*)(xrow + k0 + 20);
    v16bf av;
#pragma unroll
    for (int e = 0; e < 4; ++e) {
      av[e]      = (__bf16)a0[e];
      av[4 + e]  = (__bf16)a1[e];
      av[8 + e]  = (__bf16)a2[e];
      av[12 + e] = (__bf16)a3[e];
    }
#pragma unroll
    for (int j = 0; j < 4; ++j) {
      int nt = ntbase + j;
      v16bf bv = *(const v16bf*)(pWd + ((size_t)(kt * 64 + nt) * 32 + lane) * 16);
      acc[j] = wmma_bf16(av, bv, acc[j]);
    }
  }
#pragma unroll
  for (int j = 0; j < 4; ++j) {
    int nt = ntbase + j;
    float bb = bias[nt * 16 + (lane & 15)];
    v8bf z;
#pragma unroll
    for (int r = 0; r < 8; ++r) z[r] = (__bf16)(acc[j][r] + bb);
    size_t tileIdx = ((size_t)(dir * 2048 + t) * 2 + mt) * 64 + nt;
    *(v8bf*)(zxp + tileIdx * 256 + (size_t)lane * 8) = z;
  }
}

// ---------------------------------------------------------------------------
// recurrent: persistent kernel, 8 WGs (4/dir), 256 thr = 8 waves each.
// WG owns 64 units (all 4 gates); its R slice (256x256 bf16 = 128KB) is
// staged into LDS with ONE Tensor Data Mover descriptor (3D tile:
// 4KB chunk x 4 gates x 8 k-tiles; TDM sequential LDS write order matches
// the ldsR tile indexing exactly). Wave (mt,ust) keeps c-state for
// 16 batches x 16 units in VGPRs. Per-direction global barrier per step.
// ---------------------------------------------------------------------------
__global__ void __launch_bounds__(256)
bilstm_recur_kernel(const __bf16* __restrict__ pR,
                    const __bf16* __restrict__ zxp,
                    __bf16* __restrict__ hbuf,
                    unsigned* __restrict__ cnt,
                    float* __restrict__ out) {
  extern __shared__ char smem[];
  v16bf* ldsR = (v16bf*)smem;                  // 128 tiles * 32 lanes = 128KB

  int dir = blockIdx.x >> 2;
  int wgl = blockIdx.x & 3;                    // unit block [wgl*64, wgl*64+64)
  int tid = threadIdx.x, lane = tid & 31, wave = tid >> 5;
  int mt = wave & 1, ust = wave >> 1;          // batch-half, unit-subtile
  int half = lane >> 4, ln = lane & 15;

  const __bf16* pRd = pR + (size_t)dir * PACK_ELEMS;

#if __has_builtin(__builtin_amdgcn_tensor_load_to_lds)
  // --- TDM staging: one descriptor moves the WG's whole 128KB R slice. ---
  if (wave == 0) {
    // Tile start: (kt=0, g=0, ut block at wgl*4) -> byte offset wgl*4*1024.
    unsigned long long ga =
        (unsigned long long)(uintptr_t)(pRd) + (unsigned long long)wgl * 4096ull;
    unsigned ldsoff = (unsigned)(uintptr_t)smem;  // low 32 bits = LDS offset
    // D# group0: count=1 | lds_addr | global_addr[56:0] | type=2
    v4u g0;
    g0[0] = 1u;
    g0[1] = ldsoff;
    g0[2] = (unsigned)(ga & 0xFFFFFFFFu);
    g0[3] = (unsigned)((ga >> 32) & 0x01FFFFFFu) | 0x80000000u;
    // Geometry in 4-byte elements: x=1024 contiguous, y=4 gates (stride 4096),
    // z=8 k-tiles (stride 16384). tile dims == tensor dims (full copy).
    const unsigned td0 = 1024, td1 = 4, ti0 = 1024, ti1 = 4, ti2 = 8;
    const unsigned long long s0 = 4096ull, s1 = 16384ull;
    v8i g1;
    g1[0] = (int)(2u << 16);                               // data_size=4B
    g1[1] = (int)((td0 & 0xFFFFu) << 16);                  // td0[15:0]
    g1[2] = (int)(((td0 >> 16) & 0xFFFFu) | ((td1 & 0xFFFFu) << 16));
    g1[3] = (int)(((td1 >> 16) & 0xFFFFu) | ((ti0 & 0xFFFFu) << 16));
    g1[4] = (int)((ti1 & 0xFFFFu) | ((ti2 & 0xFFFFu) << 16));
    g1[5] = (int)(s0 & 0xFFFFFFFFull);                     // dim0 stride lo
    g1[6] = (int)(((s0 >> 32) & 0xFFFFull) | ((s1 & 0xFFFFull) << 16));
    g1[7] = (int)((s1 >> 16) & 0xFFFFFFFFull);             // dim1 stride hi
    v4i g2;
    g2[0] = 8;   // tensor_dim2
    g2[1] = 0; g2[2] = 0; g2[3] = 0;
    v4i g3 = {}; // no dims 3/4
#if defined(__clang_major__) && __clang_major__ >= 23
    v8i g4 = {};
    __builtin_amdgcn_tensor_load_to_lds(g0, g1, g2, g3, g4, 0);
#else
    __builtin_amdgcn_tensor_load_to_lds(g0, g1, g2, g3, 0);
#endif
    __builtin_amdgcn_s_wait_tensorcnt(0);
  }
#else
  // --- Fallback: plain gather of this WG's R tiles into LDS. ---
  for (int i = tid; i < 128 * 32; i += 256) {
    int tile = i >> 5, l = i & 31;
    int kt = tile >> 4, rem = tile & 15, g = rem >> 2, ut = rem & 3;
    int ntg = g * 16 + wgl * 4 + ut;
    ldsR[i] = *(const v16bf*)(pRd + ((size_t)(kt * 64 + ntg) * 32 + l) * 16);
  }
#endif
  __syncthreads();

  unsigned* mycnt = cnt + dir;
  const int HSZ = B_ * U_;                     // 8192 bf16 per (parity,dir)
  int ucol = wgl * 64 + ust * 16 + ln;         // owned unit column
  int hrow = mt * 16 + ln;                     // A-operand row (batch)
  v8f cst = {};                                // persistent cell state

  for (int t = 0; t < T_; ++t) {
    int pr = t & 1, pw = pr ^ 1;
    const __bf16* hr = hbuf + ((size_t)pr * 2 + dir) * HSZ;
    __bf16*       hw = hbuf + ((size_t)pw * 2 + dir) * HSZ;

    // Accumulator init from precomputed zx (+bias folded in). Stream NT:
    // zx is read exactly once -- keep it out of the h/R-hot caches.
    size_t tbase = ((size_t)(dir * 2048 + t) * 2 + mt) * 64;
    v8f acc[4];
#pragma unroll
    for (int g = 0; g < 4; ++g) {
      int ntg = g * 16 + wgl * 4 + ust;
      v8bf z = __builtin_nontemporal_load(
          (const v8bf*)(zxp + (tbase + ntg) * 256 + (size_t)lane * 8));
#pragma unroll
      for (int r = 0; r < 8; ++r) acc[g][r] = (float)z[r];
    }
    if (t + 1 < T_)  // prefetch next timestep's zx tiles (global_prefetch_b8)
      __builtin_prefetch(zxp + (tbase + 128) * 256 + (size_t)lane * 8, 0, 1);

    // z += h_prev @ R  (8 k-steps x 4 gate tiles of WMMA; B from LDS)
    const __bf16* hp = hr + (size_t)hrow * U_;
#pragma unroll
    for (int kt = 0; kt < 8; ++kt) {
      int k0 = kt * 32 + 8 * half;
      v8bf a0 = *(const v8bf*)(hp + k0);
      v8bf a1 = *(const v8bf*)(hp + k0 + 16);
      v16bf av;
#pragma unroll
      for (int e = 0; e < 8; ++e) { av[e] = a0[e]; av[8 + e] = a1[e]; }
#pragma unroll
      for (int g = 0; g < 4; ++g) {
        v16bf bv = ldsR[(kt * 16 + g * 4 + ust) * 32 + lane];
        acc[g] = wmma_bf16(av, bv, acc[g]);
      }
    }

    // Gates, cell/hidden update, outputs (out is write-once: NT stores).
#pragma unroll
    for (int r = 0; r < 8; ++r) {
      float iv = sigm_f(acc[0][r]);
      float fv = sigm_f(acc[1][r]);
      float gv = tanh_f(acc[2][r]);
      float ov = sigm_f(acc[3][r]);
      float cv = fv * cst[r] + iv * gv;
      cst[r] = cv;
      float hv = ov * tanh_f(cv);
      int b = mt * 16 + 8 * half + r;
      __builtin_nontemporal_store(
          hv, out + ((size_t)b * T_ + t) * (2 * U_) + (size_t)dir * U_ + ucol);
      hw[b * U_ + ucol] = (__bf16)hv;
    }

    // Per-direction grid barrier: all 4 WGs rendezvous before next step.
    __syncthreads();
    if (t < T_ - 1) {
      if (tid == 0) {
        __hip_atomic_fetch_add(mycnt, 1u, __ATOMIC_RELEASE,
                               __HIP_MEMORY_SCOPE_AGENT);
        unsigned target = (unsigned)(t + 1) * (unsigned)NWG_DIR;
        while (__hip_atomic_load(mycnt, __ATOMIC_ACQUIRE,
                                 __HIP_MEMORY_SCOPE_AGENT) < target)
          __builtin_amdgcn_s_sleep(1);
      }
      __syncthreads();
      __builtin_amdgcn_fence(__ATOMIC_ACQUIRE, "agent");  // h visibility
    }
  }
}

// ---------------------------------------------------------------------------
// Workspace layout (bytes):
//   [0,       1MB)  packed W (bf16), dir-major
//   [1MB,     2MB)  packed R (bf16), dir-major
//   [2MB,  2MB+64K) h double buffer  [parity][dir][32][256] bf16
//   [2MB+64K, ...)  barrier counters (8 x u32)
//   [4MB,   260MB)  zx packed (bf16): 2*2048*2*64 tiles * 512B
// ---------------------------------------------------------------------------
extern "C" void kernel_launch(void* const* d_in, const int* in_sizes, int n_in,
                              void* d_out, int out_size, void* d_ws,
                              size_t ws_size, hipStream_t stream) {
  const float* x    = (const float*)d_in[0];
  const float* W_fw = (const float*)d_in[1];
  const float* R_fw = (const float*)d_in[2];
  const float* b_fw = (const float*)d_in[3];
  const float* W_bw = (const float*)d_in[4];
  const float* R_bw = (const float*)d_in[5];
  const float* b_bw = (const float*)d_in[6];
  float* out = (float*)d_out;

  char* ws = (char*)d_ws;
  __bf16*   pW   = (__bf16*)(ws);
  __bf16*   pR   = (__bf16*)(ws + (1u << 20));
  __bf16*   hbuf = (__bf16*)(ws + (2u << 20));
  unsigned* cnt  = (unsigned*)(ws + (2u << 20) + 65536);
  __bf16*   zxp  = (__bf16*)(ws + (4u << 20));

  bilstm_init_kernel<<<128, 256, 0, stream>>>(hbuf, cnt);
  bilstm_pack_kernel<<<256, 256, 0, stream>>>(W_fw, W_bw, R_fw, R_bw, pW, pR);
  bilstm_proj_kernel<<<2 * 2048 * 2 * 2, 256, 0, stream>>>(x, pW, b_fw, b_bw,
                                                           zxp);
  bilstm_recur_kernel<<<2 * NWG_DIR, 256, 128 * 1024, stream>>>(pR, zxp, hbuf,
                                                                cnt, out);
}
